// TripletAttention_39625368273267
// MI455X (gfx1250) — matrix operand order repaired and checked by reading, not compiled, not run on hardware
//
#include <hip/hip_runtime.h>
#include <stdint.h>

typedef __attribute__((ext_vector_type(16))) _Float16 v16h;
typedef __attribute__((ext_vector_type(8)))  float    v8f;

#define B_   32
#define C_   256
#define H_   56
#define W_   56
#define HW_  3136
#define CHW_ 802816

// --------------------------------------------------------------------------
// Stage A1: one workgroup per (b,c) slab [56x56]. Stage slab into LDS with
// CDNA5 async global->LDS B128 loads, then compute both:
//   z1[b][2][c][w] = max/mean over h   (branch 1 ZPool)
//   z2[b][2][h][c] = max/mean over w   (branch 2 ZPool)
// --------------------------------------------------------------------------
__global__ void pool_bc(const float* __restrict__ x,
                        float* __restrict__ z1, float* __restrict__ z2) {
  __shared__ __align__(16) float tile[HW_];
  const int t  = threadIdx.x;            // 64 threads (2 waves)
  const int bc = blockIdx.x;             // b*256 + c
  const int c  = bc & 255;
  const int b  = bc >> 8;

  const uint64_t gbase = (uint64_t)(uintptr_t)(x + (size_t)bc * HW_);
  const uint32_t lbase = (uint32_t)(uintptr_t)tile;  // low 32 bits = LDS offset
  // 3136 floats = 784 float4; 12 full sweeps of 64 lanes + 16-lane tail
#pragma unroll
  for (int i = 0; i < 12; ++i) {
    uint32_t off = (uint32_t)(i * 64 + t) * 16u;
    uint32_t la  = lbase + off;
    asm volatile("global_load_async_to_lds_b128 %0, %1, %2"
                 :: "v"(la), "v"(off), "s"(gbase) : "memory");
  }
  if (t < 16) {
    uint32_t off = (uint32_t)(768 + t) * 16u;
    uint32_t la  = lbase + off;
    asm volatile("global_load_async_to_lds_b128 %0, %1, %2"
                 :: "v"(la), "v"(off), "s"(gbase) : "memory");
  }
  asm volatile("s_wait_asynccnt 0" ::: "memory");
  __syncthreads();

  if (t < 56) {
    // reduce over h at w=t  -> z1
    float mx = -3.402823466e38f, sum = 0.f;
    for (int h = 0; h < H_; ++h) {
      float v = tile[h * W_ + t];
      mx = fmaxf(mx, v); sum += v;
    }
    size_t o1 = (size_t)b * (2 * C_ * W_) + (size_t)c * W_ + t;
    z1[o1]            = mx;
    z1[o1 + C_ * W_]  = sum * (1.f / H_);

    // reduce over w at h=t  -> z2
    float mx2 = -3.402823466e38f, sum2 = 0.f;
    for (int w = 0; w < W_; ++w) {
      float v = tile[t * W_ + w];
      mx2 = fmaxf(mx2, v); sum2 += v;
    }
    size_t o2 = (size_t)b * (2 * H_ * C_) + (size_t)t * C_ + c;
    z2[o2]            = mx2;
    z2[o2 + H_ * C_]  = sum2 * (1.f / W_);
  }
}

// --------------------------------------------------------------------------
// Stage A2: one workgroup per (b,h); reduce over c (stride HW, coalesced
// across the 56 w-lanes):  z3[b][2][h][w] = max/mean over c.
// --------------------------------------------------------------------------
__global__ void pool_c(const float* __restrict__ x, float* __restrict__ z3) {
  const int t  = threadIdx.x;            // 64 threads, 56 active
  const int bh = blockIdx.x;             // b*56 + h
  const int h  = bh % 56, b = bh / 56;
  if (t >= 56) return;
  const float* p = x + (size_t)b * CHW_ + (size_t)h * W_ + t;
  float mx = -3.402823466e38f, sum = 0.f;
  for (int c4 = 0; c4 < 64; ++c4) {
    __builtin_prefetch(p + (size_t)(c4 * 4 + 32) * HW_, 0, 0);
#pragma unroll
    for (int u = 0; u < 4; ++u) {
      float v = p[(size_t)(c4 * 4 + u) * HW_];
      mx = fmaxf(mx, v); sum += v;
    }
  }
  size_t o = (size_t)b * (2 * H_ * W_) + (size_t)h * W_ + t;
  z3[o]            = mx;
  z3[o + H_ * W_]  = sum * (1.f / C_);
}

// --------------------------------------------------------------------------
// Stage B: 7x7 conv (2ch->1ch, pad 3) + BN + sigmoid on pooled map
// z[b][2][P][Q] -> sg[b][P][Q].
//
// Per-kernel-row GEMM:  Row[dp,q] = sum_{k=(c,dq)} A[dp,k] * B[k,q]
//   A (16x32 f16, built once): A[dp, k=2*dq+c] = w[c,dp,dq]; rows>=7 and
//     k>=14 are ZERO (this also nullifies any garbage in B at those k).
//   B (32x16 f16, per input row p'): lane n column = q0+n; dword j holds
//     the packed (c0,c1) f16 pair at column q0+n+4h+(j&3)-3 (h = lane/16), produced with ONE
//     ds_bpermute of each lane's packed pair (lane l holds column q0-3+l).
// One WMMA per input row feeds 7 shift-register row accumulators:
//     Yacc[6-dp] += D[dp]   (Y[p,q] = sum_dp Row[dp, p+dp-3, q])
// A wave owns (b, q-tile, p-chunk); steps over p' = pBeg-3 .. pEnd+2 and
// stores one finished 16-wide output row per step. All loads are branch-free
// (clamped address + cndmask zero), so EXEC stays all-ones for WMMA.
// --------------------------------------------------------------------------
__global__ void conv_gate(const float* __restrict__ z, float* __restrict__ sg,
                          const float* __restrict__ wt,
                          const float* __restrict__ g,  const float* __restrict__ be,
                          const float* __restrict__ mn, const float* __restrict__ vr,
                          int bi, int P, int Q, int tilesQ, int chunkP, int chunksP) {
  const int lane = threadIdx.x & 31;
  const int gw   = blockIdx.x * (blockDim.x >> 5) + (threadIdx.x >> 5);
  const int qt   = gw % tilesQ;
  int rem        = gw / tilesQ;
  const int pc   = rem % chunksP;
  const int b    = rem / chunksP;
  const int q0   = qt * 16;
  const int nsel = lane & 15;
  const bool hi  = lane >= 16;
  const int PQ   = P * Q;
  const float* zb = z + (size_t)b * 2 * PQ;

  const int pBeg = pc * chunkP;
  const int pEnd = (pBeg + chunkP < P) ? (pBeg + chunkP) : P;

  // BN constants (scalar)
  const float scl = rsqrtf(vr[bi] + 1e-5f) * g[bi];
  const float mnn = mn[bi];
  const float bet = be[bi];

  // ---- A: weights, built once.  A[m,k] with k = 2*dq + c ----
  v16h av;
#pragma unroll
  for (int j = 0; j < 8; ++j) {
    const int k0 = ((j < 4) ? (2 * j) : (16 + 2 * (j - 4))) + (hi ? 8 : 0);
    float a0 = 0.f, a1 = 0.f;
    if (nsel < 7 && k0 < 14) {
      a0 = wt[(size_t)((k0 & 1) * 49 + nsel * 7 + (k0 >> 1))];
      a1 = wt[(size_t)(((k0 + 1) & 1) * 49 + nsel * 7 + ((k0 + 1) >> 1))];
    }
    av[2 * j]     = (_Float16)a0;
    av[2 * j + 1] = (_Float16)a1;
  }

  float Yacc[7] = {0.f, 0.f, 0.f, 0.f, 0.f, 0.f, 0.f};

  const int col  = q0 - 3 + lane;                 // this lane's input column
  const int colc = col < 0 ? 0 : (col >= Q ? Q - 1 : col);
  const bool colv = (col >= 0) && (col < Q);
  const bool doSt = (lane < 16) && (q0 + nsel < Q);
  float* op = sg + (size_t)b * PQ + (size_t)pBeg * Q + q0 + nsel;  // running out ptr

  for (int pp = pBeg - 3; pp <= pEnd + 2; ++pp) {
    // ---- branch-free row fetch: two channel values at one column ----
    const bool rowv = (pp >= 0) && (pp < P);
    const int  ppc  = pp < 0 ? 0 : (pp >= P ? P - 1 : pp);
    const float* r0 = zb + (size_t)ppc * Q + colc;
    __builtin_prefetch(r0 + Q, 0, 0);             // next input row, c=0
    __builtin_prefetch(r0 + PQ + Q, 0, 0);        // next input row, c=1
    float v0 = r0[0];
    float v1 = r0[PQ];
    const bool inb = rowv && colv;
    v0 = inb ? v0 : 0.f;
    v1 = inb ? v1 : 0.f;

    union { _Float16 h[2]; int d; } pk;
    pk.h[0] = (_Float16)v0;                        // c=0
    pk.h[1] = (_Float16)v1;                        // c=1

    // ---- B: 7 bpermutes distribute the sliding window ----
    union { v16h h; int d[8]; } bu;
#pragma unroll
    for (int j = 0; j < 7; ++j) bu.d[j] = __shfl(pk.d, nsel + (j & 3) + (hi ? 4 : 0), 32);
    bu.d[7] = 0;

    v8f zero = {};
    v8f acc = __builtin_amdgcn_wmma_f32_16x16x32_f16(
        /*neg_a=*/false, av, /*neg_b=*/false, bu.h,
        /*c_mod=*/(short)0, zero, /*reuse_a=*/false, /*reuse_b=*/false);

    // Yacc[6-dp] += Row[dp]   (row m of D lives in acc[m], lanes 0-15)
    Yacc[0] += acc[6]; Yacc[1] += acc[5]; Yacc[2] += acc[4];
    Yacc[3] += acc[3]; Yacc[4] += acc[2]; Yacc[5] += acc[1];
    Yacc[6] += acc[0];

    // ---- finalize output row pp-3 (fast sigmoid: v_exp + v_rcp) ----
    const int row = pp - 3;
    if (row >= pBeg && row < pEnd) {
      if (doSt) {
        const float yy = (Yacc[0] - mnn) * scl + bet;
        *op = __builtin_amdgcn_rcpf(1.f + __expf(-yy));
      }
      op += Q;
    }

    // shift accumulator ring
    Yacc[0] = Yacc[1]; Yacc[1] = Yacc[2]; Yacc[2] = Yacc[3];
    Yacc[3] = Yacc[4]; Yacc[4] = Yacc[5]; Yacc[5] = Yacc[6];
    Yacc[6] = 0.f;
  }
}

// --------------------------------------------------------------------------
// Stage C: out = x * (s1[b,c,w] + s2[b,h,c] + s3[b,h,w]) / 3, float4-wide.
// --------------------------------------------------------------------------
__global__ void fuse_out(const float* __restrict__ x,
                         const float* __restrict__ s1, const float* __restrict__ s2,
                         const float* __restrict__ s3, float* __restrict__ out) {
  const int i4 = blockIdx.x * blockDim.x + threadIdx.x;  // 6,422,528 total
  const int w4 = i4 % 14;
  int rem      = i4 / 14;
  const int h  = rem % 56;
  rem         /= 56;
  const int c  = rem & 255;
  const int b  = rem >> 8;
  const size_t e = (size_t)i4 * 4;

  const float4 xv = *(const float4*)(x + e);
  const float4 a1 = *(const float4*)(s1 + (size_t)b * (C_ * W_) + c * W_ + w4 * 4);
  const float  a2 = s2[(size_t)b * (H_ * C_) + h * C_ + c];
  const float4 a3 = *(const float4*)(s3 + (size_t)b * (H_ * W_) + h * W_ + w4 * 4);

  float4 r;
  r.x = xv.x * (a1.x + a2 + a3.x) * (1.f / 3.f);
  r.y = xv.y * (a1.y + a2 + a3.y) * (1.f / 3.f);
  r.z = xv.z * (a1.z + a2 + a3.z) * (1.f / 3.f);
  r.w = xv.w * (a1.w + a2 + a3.w) * (1.f / 3.f);
  *(float4*)(out + e) = r;
}

// --------------------------------------------------------------------------
extern "C" void kernel_launch(void* const* d_in, const int* in_sizes, int n_in,
                              void* d_out, int out_size, void* d_ws, size_t ws_size,
                              hipStream_t stream) {
  const float* x    = (const float*)d_in[0];
  const float* w_ch = (const float*)d_in[1];
  const float* w_cw = (const float*)d_in[2];
  const float* w_hw = (const float*)d_in[3];
  const float* gm   = (const float*)d_in[4];
  const float* bt   = (const float*)d_in[5];
  const float* mn   = (const float*)d_in[6];
  const float* vr   = (const float*)d_in[7];
  float* out = (float*)d_out;

  // workspace carve-out (floats): 3,053,568 floats = ~11.65 MB
  float* ws = (float*)d_ws;
  float* z1 = ws;                       // [32][2][256][56] = 917504
  float* z2 = z1 + 917504;              // [32][2][56][256] = 917504
  float* z3 = z2 + 917504;              // [32][2][56][56]  = 200704
  float* s1 = z3 + 200704;              // [32][256][56]    = 458752
  float* s2 = s1 + 458752;              // [32][56][256]    = 458752
  float* s3 = s2 + 458752;              // [32][56][56]     = 100352

  // Stage A: pooled maps
  pool_bc<<<B_ * C_, 64, 0, stream>>>(x, z1, z2);          // 8192 blocks
  pool_c <<<B_ * H_, 64, 0, stream>>>(x, z3);              // 1792 blocks

  // Stage B: WMMA conv + BN + sigmoid per branch
  //   branch1: P=256(c), Q=56(w),  tilesQ=4,  chunkP=16 x16 -> 2048 waves
  //   branch2: P=56(h),  Q=256(c), tilesQ=16, chunkP=14 x4  -> 2048 waves
  //   branch3: P=56(h),  Q=56(w),  tilesQ=4,  chunkP=14 x4  -> 512 waves
  conv_gate<<<512, 128, 0, stream>>>(z1, s1, w_ch, gm, bt, mn, vr, 0, 256, 56, 4, 16, 16);
  conv_gate<<<512, 128, 0, stream>>>(z2, s2, w_cw, gm, bt, mn, vr, 1, 56, 256, 16, 14, 4);
  conv_gate<<<128, 128, 0, stream>>>(z3, s3, w_hw, gm, bt, mn, vr, 2, 56, 56, 4, 14, 4);

  // Stage C: fused gating, float4-wide (25,690,112 / 4 = 6,422,528 lanes)
  fuse_out<<<25088, 256, 0, stream>>>(x, s1, s2, s3, out);
}
